// RobustIMUEncoder_14259291423005
// MI455X (gfx1250) — matrix operand order repaired and checked by reading, not compiled, not run on hardware
//
#include <hip/hip_runtime.h>
#include <hip/hip_fp16.h>
#include <hip/hip_bf16.h>

// ---------------------------------------------------------------------------
// RobustIMUEncoder for MI455X (gfx1250, wave32, WMMA).
//
// ~0.15 TFLOP total, dominated by sequential LSTM recurrent GEMMs.
// - All matmuls via v_wmma_f32_16x16x32_f16 (f16 in, f32 accumulate).
// - Weights pre-swizzled into B-fragment order: each lane fetches its 16
//   operand halves as two contiguous global b128 loads (L2-resident).
// - Scan keeps h/c/gates in LDS (60 KB/block << 320 KB WGP pool); the A
//   operand [h_prev | x_t] is staged in LDS and read as two ds b128 loads
//   per 16x32 fragment. One block = 16 batch rows; 8 waves cover all 512
//   gate columns per timestep.
// - The four independent layer-1 scans run in ONE launch (gridDim.y = 4),
//   same for layer-2: 4x the blocks on the serialized critical path.
// ---------------------------------------------------------------------------

typedef __attribute__((ext_vector_type(16))) _Float16 v16h;
typedef __attribute__((ext_vector_type(8)))  float    v8f;

union Frag32B { uint4 q[2]; v16h v; };

// A-fragment (16x32 f16, ISA 7.12.2): lanes 0-15 hold M=lane, K = 0..7,16..23;
// lanes 16-31 hold M=lane-16, K = 8..15,24..31. Two 16B LDS reads per lane.
__device__ __forceinline__ v16h load_a_frag(const _Float16* a_tile, int kpad,
                                            int kk, int lane) {
  int row = lane & 15;
  int k0  = kk * 32 + ((lane >> 4) << 3);      // +0 or +8
  const uint4* p = (const uint4*)(a_tile + row * kpad + k0);
  Frag32B f; f.q[0] = p[0]; f.q[1] = p[2];     // k0..k0+7, k0+16..k0+23
  return f.v;
}

// B-fragment: pre-swizzled in global memory, fragment-major:
// idx = ((nt*NC + kk)*32 + lane)*16 + j  ->  32 contiguous bytes per lane.
__device__ __forceinline__ v16h load_b_frag(const _Float16* frags, int nc,
                                            int nt, int kk, int lane) {
  const uint4* p = (const uint4*)(frags + (((size_t)(nt * nc + kk)) * 32 + lane) * 16);
  Frag32B f; f.q[0] = p[0]; f.q[1] = p[1];
  return f.v;
}

__device__ __forceinline__ v8f wmma_step(v16h a, v16h b, v8f acc) {
  return __builtin_amdgcn_wmma_f32_16x16x32_f16(
      /*neg_a=*/false, a, /*neg_b=*/false, b,
      /*c_mod=*/(short)0, acc, /*reuse_a=*/false, /*reuse_b=*/false);
}

// ---------------------------------------------------------------------------
// Weight packer: builds B = W'^T in WMMA fragment order (f16), where
// W'(n,k) = (k < K1 ? w1[n*K1+k] : (k-K1 < Din ? w2[n*Din+(k-K1)] : 0)).
// For the LSTM scan w1=Whh (K1=128), w2=Wih -> combined [h|x] GEMM.
// For plain linears w1=null,K1=0.
// ---------------------------------------------------------------------------
__global__ void pack_b_frags(const float* w1, int K1, const float* w2, int Din,
                             int Kpad, int Nout, _Float16* frags) {
  int idx = blockIdx.x * blockDim.x + threadIdx.x;
  int total = Kpad * Nout;
  if (idx >= total) return;
  int NC   = Kpad >> 5;
  int j    = idx & 15;
  int lane = (idx >> 4) & 31;
  int rest = idx >> 9;
  int kk   = rest % NC;
  int nt   = rest / NC;
  int k = kk * 32 + ((j < 8) ? (((lane >> 4) << 3) + j) : (16 + ((lane >> 4) << 3) + (j - 8)));
  int n = nt * 16 + (lane & 15);
  float v = 0.f;
  if (k < K1) {
    v = w1[(size_t)n * K1 + k];
  } else {
    int kd = k - K1;
    if (kd < Din) v = w2[(size_t)n * Din + kd];
  }
  frags[idx] = (_Float16)v;
}

__global__ void add_bias2(const float* a, const float* b, float* o, int n) {
  int i = blockIdx.x * blockDim.x + threadIdx.x;
  if (i < n) o[i] = a[i] + b[i];
}

// ---------------------------------------------------------------------------
// LSTM scan: blockIdx.y selects one of 4 independent scan jobs (fwd/bwd x
// raw/fixed). One block = 16 batch rows, loops over time internally.
// Per step: gates[16x512] = [h|x_t][16xKPAD] @ Wcomb^T (WMMA), + bias,
// then the LSTM cell elementwise in f32.
// ---------------------------------------------------------------------------
#define HIDN 128
#define G4   512

struct ScanJob {
  const float*    xf;        // f32 input (layer 1) or null
  const _Float16* xh;        // f16 input (layer 2) or null
  const _Float16* frags;     // packed [h|x] weights
  const float*    bias;      // bih+bhh
  _Float16*       out_seq;   // per-step h (layer 1) or null
  float*          out_final; // final h (layer 2) or null
  int Din, DinPad, Lseq, reverse, out_col, fin_stride, fin_col;
};

__global__ __launch_bounds__(256) void lstm_scan4(ScanJob j0, ScanJob j1,
                                                  ScanJob j2, ScanJob j3, int N) {
  ScanJob j = (blockIdx.y == 0) ? j0 : (blockIdx.y == 1) ? j1
            : (blockIdx.y == 2) ? j2 : j3;
  const int KPAD = HIDN + j.DinPad;
  const int NC   = KPAD >> 5;
  __shared__ float hbuf[16 * HIDN];
  __shared__ float cbuf[16 * HIDN];
  __shared__ __attribute__((aligned(16))) _Float16 abuf[16 * (HIDN + 256)];
  __shared__ float gates[16 * G4];

  const int tid  = threadIdx.x;
  const int lane = tid & 31;
  const int wave = tid >> 5;
  const int m0   = blockIdx.x * 16;

  for (int i = tid; i < 16 * HIDN; i += 256) { hbuf[i] = 0.f; cbuf[i] = 0.f; }
  __syncthreads();

  for (int tt = 0; tt < j.Lseq; ++tt) {
    const int t = j.reverse ? (j.Lseq - 1 - tt) : tt;

    // Prefetch next timestep's input rows (global_prefetch_b8).
    if (j.xf && tid < 16) {
      int tn = j.reverse ? (t > 0 ? t - 1 : 0) : (t + 1 < j.Lseq ? t + 1 : t);
      int m  = m0 + tid;
      if (m < N) __builtin_prefetch(&j.xf[((size_t)m * j.Lseq + tn) * j.Din], 0, 0);
    }

    // Stage A = [h_prev | x_t] into LDS as f16.
    if (j.xh) {
      // f16 sequence input: 8-half (16B) vector copies, rows are 16B aligned.
      const int rowv = j.Din >> 3;
      for (int i = tid; i < 16 * rowv; i += 256) {
        int r = i / rowv, c = i - r * rowv;
        int m = m0 + r;
        uint4 v = {0u, 0u, 0u, 0u};
        if (m < N)
          v = *(const uint4*)(j.xh + ((size_t)m * j.Lseq + t) * j.Din + c * 8);
        *(uint4*)(abuf + r * KPAD + HIDN + c * 8) = v;
      }
    } else {
      // f32 input (Din=6, padded to DinPad=32 with zeros).
      for (int i = tid; i < 16 * j.DinPad; i += 256) {
        int r = i / j.DinPad, kd = i - r * j.DinPad;
        int m = m0 + r;
        float v = (kd < j.Din && m < N)
                      ? j.xf[((size_t)m * j.Lseq + t) * j.Din + kd] : 0.f;
        abuf[r * KPAD + HIDN + kd] = (_Float16)v;
      }
    }
    for (int i = tid; i < 16 * HIDN; i += 256) {
      int r = i >> 7, k = i & 127;
      abuf[r * KPAD + k] = (_Float16)hbuf[i];
    }
    __syncthreads();

    // 8 waves x 4 tiles of 16 cols = 512 gate outputs; K accumulated by WMMA.
    for (int i2 = 0; i2 < 4; ++i2) {
      int nt = wave * 4 + i2;
      v8f acc = {0.f, 0.f, 0.f, 0.f, 0.f, 0.f, 0.f, 0.f};
      for (int kk = 0; kk < NC; ++kk) {
        v16h a = load_a_frag(abuf, KPAD, kk, lane);
        v16h b = load_b_frag(j.frags, NC, nt, kk, lane);
        acc = wmma_step(a, b, acc);
      }
      int ncol = nt * 16 + (lane & 15);
      float bb = j.bias[ncol];
#pragma unroll
      for (int r = 0; r < 8; ++r) {
        int row = (lane < 16) ? r : (r + 8);
        gates[row * G4 + ncol] = acc[r] + bb;
      }
    }
    __syncthreads();

    // LSTM cell (f32): i,f,g,o -> c,h.
    for (int i = tid; i < 16 * HIDN; i += 256) {
      int r = i >> 7, jj = i & 127;
      float iv = gates[r * G4 + jj];
      float fv = gates[r * G4 + 128 + jj];
      float gv = gates[r * G4 + 256 + jj];
      float ov = gates[r * G4 + 384 + jj];
      float si = 1.f / (1.f + __expf(-iv));
      float sf = 1.f / (1.f + __expf(-fv));
      float so = 1.f / (1.f + __expf(-ov));
      float c  = sf * cbuf[i] + si * tanhf(gv);
      float h  = so * tanhf(c);
      cbuf[i] = c;
      hbuf[i] = h;
      int m = m0 + r;
      if (j.out_seq && m < N)
        j.out_seq[((size_t)m * j.Lseq + t) * 256 + j.out_col + jj] = (_Float16)h;
    }
    __syncthreads();
  }

  if (j.out_final) {
    for (int i = tid; i < 16 * HIDN; i += 256) {
      int r = i >> 7, jj = i & 127;
      int m = m0 + r;
      if (m < N) j.out_final[(size_t)m * j.fin_stride + j.fin_col + jj] = hbuf[i];
    }
  }
}

// ---------------------------------------------------------------------------
// Generic WMMA GEMM: C[m, Coffs+n] = A[m,:K] @ W^T (+bias). A is f32, staged
// to LDS f16; W pre-packed by pack_b_frags. Block covers 16 rows x all cols.
// ---------------------------------------------------------------------------
__global__ __launch_bounds__(256) void gemm16(
    const float* A, int Astride, int M, int K, int Kpad,
    const _Float16* frags, const float* bias,
    float* C, int Cstride, int Coffs, int tilesPerWave) {
  __shared__ __attribute__((aligned(16))) _Float16 a_tile[16 * 256];
  const int tid = threadIdx.x, lane = tid & 31, wave = tid >> 5;
  const int m0 = blockIdx.x * 16;

  for (int i = tid; i < 16 * Kpad; i += 256) {
    int r = i / Kpad, k = i - r * Kpad;
    int m = m0 + r;
    float v = (m < M && k < K) ? A[(size_t)m * Astride + k] : 0.f;
    a_tile[i] = (_Float16)v;
  }
  __syncthreads();

  const int NC = Kpad >> 5;
  for (int i2 = 0; i2 < tilesPerWave; ++i2) {
    int nt = wave * tilesPerWave + i2;
    v8f acc = {0.f, 0.f, 0.f, 0.f, 0.f, 0.f, 0.f, 0.f};
    for (int kk = 0; kk < NC; ++kk) {
      v16h a = load_a_frag(a_tile, Kpad, kk, lane);
      v16h b = load_b_frag(frags, NC, nt, kk, lane);
      acc = wmma_step(a, b, acc);
    }
    int ncol = nt * 16 + (lane & 15);
    float bb = bias ? bias[ncol] : 0.f;
#pragma unroll
    for (int r = 0; r < 8; ++r) {
      int row = (lane < 16) ? r : (r + 8);
      int m = m0 + row;
      if (m < M) C[(size_t)m * Cstride + Coffs + ncol] = acc[r] + bb;
    }
  }
}

// ---------------------------------------------------------------------------
// Small VALU kernels: stats, resample, 3-token attention, token mean, LN+ReLU.
// ---------------------------------------------------------------------------
__global__ void stats_kernel(const float* x, float* stats, int N, int Lseq, int D) {
  int idx = blockIdx.x * blockDim.x + threadIdx.x;
  if (idx >= N * D) return;
  int n = idx / D, d = idx - n * D;
  const float* p = x + (size_t)n * Lseq * D + d;
  float s = 0.f, s2 = 0.f, mn = 1e30f, mx = -1e30f;
  for (int t = 0; t < Lseq; ++t) {
    float v = p[(size_t)t * D];
    s += v; s2 += v * v;
    mn = fminf(mn, v); mx = fmaxf(mx, v);
  }
  float mean = s / (float)Lseq;
  float var  = fmaxf((s2 - (float)Lseq * mean * mean) / (float)(Lseq - 1), 0.f);
  float* o = stats + (size_t)n * 4 * D;
  o[0 * D + d] = mean;
  o[1 * D + d] = sqrtf(var);
  o[2 * D + d] = mn;
  o[3 * D + d] = mx;
}

__global__ void resample_kernel(const float* x, float* xr, int N, int L, int Tgt, int D) {
  int idx = blockIdx.x * blockDim.x + threadIdx.x;
  if (idx >= N * Tgt * D) return;
  int d = idx % D;
  int t = (idx / D) % Tgt;
  int n = idx / (D * Tgt);
  float pos = (float)t * (float)(L - 1) / (float)(Tgt - 1);
  int lo = (int)floorf(pos);
  int hi = lo + 1 < L ? lo + 1 : L - 1;
  float fr = pos - (float)lo;
  xr[idx] = x[((size_t)n * L + lo) * D + d] * (1.f - fr) +
            x[((size_t)n * L + hi) * D + d] * fr;
}

__global__ void attn3_kernel(const float* qkv, float* o, int N) {
  int id = blockIdx.x * blockDim.x + threadIdx.x;
  if (id >= N * 4) return;
  int n = id >> 2, h = id & 3;
  const float* base = qkv + (size_t)n * 3 * 768 + h * 64;
  float s[3][3];
  for (int j = 0; j < 3; ++j)
    for (int kk = 0; kk < 3; ++kk) {
      const float* q = base + j * 768;
      const float* k = base + kk * 768 + 256;
      float acc = 0.f;
      for (int d = 0; d < 64; ++d) acc += q[d] * k[d];
      s[j][kk] = acc * 0.125f;  // 1/sqrt(64)
    }
  for (int j = 0; j < 3; ++j) {
    float mx = fmaxf(s[j][0], fmaxf(s[j][1], s[j][2]));
    float e0 = __expf(s[j][0] - mx), e1 = __expf(s[j][1] - mx), e2 = __expf(s[j][2] - mx);
    float inv = 1.f / (e0 + e1 + e2);
    s[j][0] = e0 * inv; s[j][1] = e1 * inv; s[j][2] = e2 * inv;
  }
  float* ob = o + (size_t)n * 3 * 256 + h * 64;
  for (int j = 0; j < 3; ++j)
    for (int d = 0; d < 64; ++d)
      ob[j * 256 + d] = s[j][0] * base[0 * 768 + 512 + d] +
                        s[j][1] * base[1 * 768 + 512 + d] +
                        s[j][2] * base[2 * 768 + 512 + d];
}

__global__ void mean3_kernel(const float* ao, float* feat, int N) {
  int idx = blockIdx.x * blockDim.x + threadIdx.x;
  if (idx >= N * 256) return;
  int n = idx >> 8, c = idx & 255;
  const float* p = ao + (size_t)n * 768;
  feat[idx] = (p[c] + p[256 + c] + p[512 + c]) * (1.f / 3.f);
}

__global__ __launch_bounds__(256) void ln_relu_kernel(
    const float* y, const float* g, const float* b, float* out, int N) {
  __shared__ float red[256];
  int row = blockIdx.x, tid = threadIdx.x;
  float v = y[(size_t)row * 256 + tid];
  red[tid] = v; __syncthreads();
  for (int s = 128; s > 0; s >>= 1) { if (tid < s) red[tid] += red[tid + s]; __syncthreads(); }
  float mean = red[0] * (1.f / 256.f); __syncthreads();
  float dv = v - mean;
  red[tid] = dv * dv; __syncthreads();
  for (int s = 128; s > 0; s >>= 1) { if (tid < s) red[tid] += red[tid + s]; __syncthreads(); }
  float var = red[0] * (1.f / 256.f);
  float o = dv * rsqrtf(var + 1e-5f) * g[tid] + b[tid];
  out[(size_t)row * 256 + tid] = fmaxf(o, 0.f);
}

// ---------------------------------------------------------------------------
// Host: workspace plan + launch sequence (all on `stream`, graph-safe).
// ---------------------------------------------------------------------------
extern "C" void kernel_launch(void* const* d_in, const int* in_sizes, int n_in,
                              void* d_out, int out_size, void* d_ws, size_t ws_size,
                              hipStream_t stream) {
  (void)in_sizes; (void)n_in; (void)out_size; (void)ws_size;
  const int N = 24 * 41;      // 984 sequences
  const int L = 100, TGT = 50, D = 6;
  const int MB  = (N + 15) / 16;        // 62 row-tiles
  const int M3  = N * 3;                // 2952 token rows
  const int MB3 = (M3 + 15) / 16;       // 185

  const float* imu = (const float*)d_in[0];
  // jax pytree flattening: dict keys sorted -> per layer: bwd(Wih,Whh,bih,bhh), fwd(...)
  auto lstm_ptr = [&](int base, int layer, int isFwd, int which) {
    return (const float*)d_in[base + layer * 8 + (isFwd ? 4 : 0) + which];
  };
  const int RAW = 1, FIX = 17;
  const float* stats_W    = (const float*)d_in[33];
  const float* stats_b    = (const float*)d_in[34];
  const float* attn_in_W  = (const float*)d_in[35];
  const float* attn_in_b  = (const float*)d_in[36];
  const float* attn_out_W = (const float*)d_in[37];
  const float* attn_out_b = (const float*)d_in[38];
  const float* proj_W     = (const float*)d_in[39];
  const float* proj_b     = (const float*)d_in[40];
  const float* ln_g       = (const float*)d_in[41];
  const float* ln_b       = (const float*)d_in[42];

  // Bump allocator over d_ws.
  size_t off = 0;
  auto alloc = [&](size_t bytes) -> char* {
    char* p = (char*)d_ws + off;
    off += (bytes + 255) & ~(size_t)255;
    return p;
  };

  float*     xr       = (float*)alloc((size_t)N * TGT * D * 4);
  float*     statsBuf = (float*)alloc((size_t)N * 24 * 4);
  float*     tok      = (float*)alloc((size_t)N * 3 * 256 * 4);
  _Float16*  seq_fix  = (_Float16*)alloc((size_t)N * TGT * 256 * 2);
  // Scan fragments: 4x layer1 (Kpad=160) then 4x layer2 (Kpad=384);
  // order within each group: rawF, rawB, fixF, fixB.
  _Float16*  scanF[8];
  float*     scanB[8];
  for (int i = 0; i < 4; ++i) scanF[i]     = (_Float16*)alloc((size_t)160 * 512 * 2);
  for (int i = 0; i < 4; ++i) scanF[4 + i] = (_Float16*)alloc((size_t)384 * 512 * 2);
  for (int i = 0; i < 8; ++i) scanB[i]     = (float*)alloc(512 * 4);
  _Float16* statsFr = (_Float16*)alloc((size_t)32 * 256 * 2);
  _Float16* inFr    = (_Float16*)alloc((size_t)256 * 768 * 2);
  _Float16* outFr   = (_Float16*)alloc((size_t)256 * 256 * 2);
  _Float16* projFr  = (_Float16*)alloc((size_t)256 * 256 * 2);
  // Big raw sequence buffer; late-phase buffers alias it once layer-2 scans finish.
  char*     seqRawRegion = alloc((size_t)N * L * 256 * 2);   // 50.4 MB
  _Float16* seq_raw = (_Float16*)seqRawRegion;
  float*    qkv  = (float*)(seqRawRegion);                                 // 9.1 MB
  float*    obuf = (float*)(seqRawRegion + (((size_t)M3 * 768 * 4 + 255) & ~(size_t)255));
  float*    ao   = (float*)((char*)obuf + (((size_t)M3 * 256 * 4 + 255) & ~(size_t)255));
  float*    feat = (float*)((char*)ao   + (((size_t)M3 * 256 * 4 + 255) & ~(size_t)255));
  float*    ybuf = (float*)((char*)feat + (((size_t)N  * 256 * 4 + 255) & ~(size_t)255));

  // ---- Pack weights into WMMA fragment order + combine LSTM biases ----
  // instance order: 0 rawL1f, 1 rawL1b, 2 fixL1f, 3 fixL1b,
  //                 4 rawL2f, 5 rawL2b, 6 fixL2f, 7 fixL2b
  struct InstDesc { int base, layer, fwd; };
  InstDesc inst[8] = {
      {RAW, 0, 1}, {RAW, 0, 0}, {FIX, 0, 1}, {FIX, 0, 0},
      {RAW, 1, 1}, {RAW, 1, 0}, {FIX, 1, 1}, {FIX, 1, 0}};
  for (int i = 0; i < 8; ++i) {
    int layer = inst[i].layer;
    int Din   = layer == 0 ? 6 : 256;
    int Kpad  = layer == 0 ? 160 : 384;
    const float* Wih = lstm_ptr(inst[i].base, layer, inst[i].fwd, 0);
    const float* Whh = lstm_ptr(inst[i].base, layer, inst[i].fwd, 1);
    const float* bih = lstm_ptr(inst[i].base, layer, inst[i].fwd, 2);
    const float* bhh = lstm_ptr(inst[i].base, layer, inst[i].fwd, 3);
    int total = Kpad * 512;
    pack_b_frags<<<(total + 255) / 256, 256, 0, stream>>>(Whh, 128, Wih, Din, Kpad, 512, scanF[i]);
    add_bias2<<<2, 256, 0, stream>>>(bih, bhh, scanB[i], 512);
  }
  pack_b_frags<<<(32 * 256 + 255) / 256, 256, 0, stream>>>(nullptr, 0, stats_W, 24, 32, 256, statsFr);
  pack_b_frags<<<(256 * 768 + 255) / 256, 256, 0, stream>>>(nullptr, 0, attn_in_W, 256, 256, 768, inFr);
  pack_b_frags<<<(256 * 256 + 255) / 256, 256, 0, stream>>>(nullptr, 0, attn_out_W, 256, 256, 256, outFr);
  pack_b_frags<<<(256 * 256 + 255) / 256, 256, 0, stream>>>(nullptr, 0, proj_W, 256, 256, 256, projFr);

  // ---- Stats + resample ----
  stats_kernel<<<(N * D + 255) / 256, 256, 0, stream>>>(imu, statsBuf, N, L, D);
  resample_kernel<<<(N * TGT * D + 255) / 256, 256, 0, stream>>>(imu, xr, N, L, TGT, D);

  // ---- BiLSTM scans: 4 independent layer-1 scans in one launch ----
  auto mkjob = [](const float* xf, const _Float16* xh, const _Float16* fr,
                  const float* bi, _Float16* oseq, float* ofin,
                  int Din, int DinPad, int Lseq, int rev, int ocol,
                  int fstride, int fcol) {
    ScanJob s;
    s.xf = xf; s.xh = xh; s.frags = fr; s.bias = bi;
    s.out_seq = oseq; s.out_final = ofin;
    s.Din = Din; s.DinPad = DinPad; s.Lseq = Lseq; s.reverse = rev;
    s.out_col = ocol; s.fin_stride = fstride; s.fin_col = fcol;
    return s;
  };
  {
    ScanJob a = mkjob(imu, nullptr, scanF[0], scanB[0], seq_raw, nullptr, 6, 32, L,   0, 0,   0, 0);
    ScanJob b = mkjob(imu, nullptr, scanF[1], scanB[1], seq_raw, nullptr, 6, 32, L,   1, 128, 0, 0);
    ScanJob c = mkjob(xr,  nullptr, scanF[2], scanB[2], seq_fix, nullptr, 6, 32, TGT, 0, 0,   0, 0);
    ScanJob d = mkjob(xr,  nullptr, scanF[3], scanB[3], seq_fix, nullptr, 6, 32, TGT, 1, 128, 0, 0);
    lstm_scan4<<<dim3(MB, 4), 256, 0, stream>>>(a, b, c, d, N);
  }
  // ---- 4 independent layer-2 scans in one launch; final h -> tokens 0/1 ----
  {
    ScanJob a = mkjob(nullptr, seq_raw, scanF[4], scanB[4], nullptr, tok,       256, 256, L,   0, 0, 768, 0);
    ScanJob b = mkjob(nullptr, seq_raw, scanF[5], scanB[5], nullptr, tok,       256, 256, L,   1, 0, 768, 128);
    ScanJob c = mkjob(nullptr, seq_fix, scanF[6], scanB[6], nullptr, tok + 256, 256, 256, TGT, 0, 0, 768, 0);
    ScanJob d = mkjob(nullptr, seq_fix, scanF[7], scanB[7], nullptr, tok + 256, 256, 256, TGT, 1, 0, 768, 128);
    lstm_scan4<<<dim3(MB, 4), 256, 0, stream>>>(a, b, c, d, N);
  }

  // ---- stats linear -> token 2 of tok ----
  gemm16<<<MB, 256, 0, stream>>>(statsBuf, 24, N, 24, 32, statsFr, stats_b, tok, 768, 512, 2);

  // ---- attention: qkv, 3x3 softmax, out-proj, token mean ----
  // NOTE: qkv aliases seq_raw (raw scans are complete at this point).
  gemm16<<<MB3, 256, 0, stream>>>(tok, 256, M3, 256, 256, inFr, attn_in_b, qkv, 768, 0, 6);
  attn3_kernel<<<(N * 4 + 255) / 256, 256, 0, stream>>>(qkv, obuf, N);
  gemm16<<<MB3, 256, 0, stream>>>(obuf, 256, M3, 256, 256, outFr, attn_out_b, ao, 256, 0, 2);
  mean3_kernel<<<(N * 256 + 255) / 256, 256, 0, stream>>>(ao, feat, N);

  // ---- projection + layernorm + relu ----
  gemm16<<<MB, 256, 0, stream>>>(feat, 256, N, 256, 256, projFr, proj_b, ybuf, 256, 0, 2);
  ln_relu_kernel<<<N, 256, 0, stream>>>(ybuf, ln_g, ln_b, (float*)d_out, N);
}
